// GATLayer_30193620090943
// MI455X (gfx1250) — matrix operand order repaired and checked by reading, not compiled, not run on hardware
//
#include <hip/hip_runtime.h>
#include <hip/hip_bf16.h>
#include <math.h>

#define N_NODES   100000
#define N_EDGES   1600000
#define IN_DIM    128
#define HEADS     4
#define OUT_DIM   16
#define HD        (HEADS * OUT_DIM)   // 64
#define NEG_SLOPE 0.2f

typedef __attribute__((ext_vector_type(16))) _Float16 v16h;
typedef __attribute__((ext_vector_type(8)))  float    v8f;

// ---------------- order-preserving float<->uint encoding for atomic max ----
__device__ __forceinline__ unsigned int encF(float f) {
    unsigned int u = __float_as_uint(f);
    return (u & 0x80000000u) ? ~u : (u | 0x80000000u);
}
__device__ __forceinline__ float decF(unsigned int u) {
    return (u & 0x80000000u) ? __uint_as_float(u ^ 0x80000000u)
                             : __uint_as_float(~u);
}
// encF(-inf) == ~0xFF800000 == 0x007FFFFF
#define ENC_NEG_INF 0x007FFFFFu

__device__ __forceinline__ float lrelu(float v) {
    return v > 0.0f ? v : v * NEG_SLOPE;
}

// ---------------- kernel 0: init out=bias, denom=0, nmax=-inf --------------
__global__ __launch_bounds__(256)
void gat_init(float* __restrict__ out, const float* __restrict__ bias,
              float* __restrict__ denom, unsigned int* __restrict__ nmax) {
    int i = blockIdx.x * blockDim.x + threadIdx.x;
    if (i < N_NODES * HD) out[i] = bias[i & (HD - 1)];
    if (i < N_NODES * HEADS) { denom[i] = 0.0f; nmax[i] = ENC_NEG_INF; }
}

// ---------------- kernel 1: h = feat @ W via V_WMMA_F32_16X16X32_F16 -------
// Block = 128 threads = 4 waves. Wave w computes the 16x16 tile for head w.
// Grid.x = N_NODES/16 row tiles.
__global__ __launch_bounds__(128)
void gat_gemm_wmma(const float* __restrict__ feat, const float* __restrict__ W,
                   float* __restrict__ h) {
    const int wave = threadIdx.x >> 5;        // 0..3  -> N tile (head)
    const int lane = threadIdx.x & 31;
    const int r    = lane & 15;
    const int hiL  = lane >> 4;               // 0 or 1
    const int row0 = blockIdx.x * 16;
    const int col0 = wave * 16;

    const float* arow = feat + (size_t)(row0 + r) * IN_DIM;
    const float* bcol = W + (col0 + r);       // column of W, stride HD

    v8f acc = {};
    #pragma unroll
    for (int ks = 0; ks < 4; ++ks) {
        const int K0  = ks * 32;
        const int kbA = K0 + hiL * 16;        // A per-lane K base (contiguous 16 block)
        const int kbB = K0 + hiL * 16;        // B per-lane K base (contiguous 16 block)

        // A tile: lane holds row (row0+r), 16 K values {kbA..kbA+15}
        float4 p0 = *(const float4*)(arow + kbA);
        float4 p1 = *(const float4*)(arow + kbA + 4);
        float4 p2 = *(const float4*)(arow + kbA + 8);
        float4 p3 = *(const float4*)(arow + kbA + 12);
        v16h a;
        a[0] = (_Float16)p0.x;  a[1] = (_Float16)p0.y;
        a[2] = (_Float16)p0.z;  a[3] = (_Float16)p0.w;
        a[4] = (_Float16)p1.x;  a[5] = (_Float16)p1.y;
        a[6] = (_Float16)p1.z;  a[7] = (_Float16)p1.w;
        a[8]  = (_Float16)p2.x; a[9]  = (_Float16)p2.y;
        a[10] = (_Float16)p2.z; a[11] = (_Float16)p2.w;
        a[12] = (_Float16)p3.x; a[13] = (_Float16)p3.y;
        a[14] = (_Float16)p3.z; a[15] = (_Float16)p3.w;

        // B tile: lane holds column (col0+r), K = kbB + j (W is tiny -> L1/L2 hits)
        v16h b;
        #pragma unroll
        for (int j = 0; j < 16; ++j)
            b[j] = (_Float16)bcol[(size_t)(kbB + j) * HD];

        acc = __builtin_amdgcn_wmma_f32_16x16x32_f16(
            /*neg_a=*/false, a, /*neg_b=*/false, b,
            /*c_mod=*/(short)0, acc, /*reuse_a=*/false, /*reuse_b=*/false);
    }

    // D layout: VGPR rr -> row rr + 8*hiL, col = r
    float* orow = h + (size_t)row0 * HD + col0;
    #pragma unroll
    for (int rr = 0; rr < 8; ++rr)
        orow[(size_t)(rr + hiL * 8) * HD + r] = acc[rr];
}

// ---------------- kernel 2: per-node attention halves el/er ---------------
__global__ __launch_bounds__(256)
void gat_scores(const float* __restrict__ h, const float* __restrict__ attn_l,
                const float* __restrict__ attn_r,
                float* __restrict__ el, float* __restrict__ er) {
    int i = blockIdx.x * blockDim.x + threadIdx.x;     // node*HEADS + head
    if (i >= N_NODES * HEADS) return;
    int node = i >> 2, head = i & 3;
    const float* hp = h + (size_t)node * HD + head * OUT_DIM;
    const float* al = attn_l + head * OUT_DIM;
    const float* ar = attn_r + head * OUT_DIM;
    float sl = 0.0f, sr = 0.0f;
    #pragma unroll
    for (int d = 0; d < OUT_DIM; ++d) {
        float v = hp[d];
        sl += v * al[d];
        sr += v * ar[d];
    }
    el[i] = sl; er[i] = sr;
}

// ---------------- kernel 3: edge logits -> segment max --------------------
__global__ __launch_bounds__(256)
void gat_edge_max(const int* __restrict__ src, const int* __restrict__ dst,
                  const float* __restrict__ el, const float* __restrict__ er,
                  unsigned int* __restrict__ nmax) {
    int e = blockIdx.x * blockDim.x + threadIdx.x;
    if (e >= N_EDGES) return;
    int s = src[e], d = dst[e];
    #pragma unroll
    for (int hh = 0; hh < HEADS; ++hh) {
        float v = lrelu(el[s * HEADS + hh] + er[d * HEADS + hh]);
        atomicMax(&nmax[d * HEADS + hh], encF(v));
    }
}

// ---------------- kernel 4: exp + segment sum (denominator) ---------------
__global__ __launch_bounds__(256)
void gat_edge_denom(const int* __restrict__ src, const int* __restrict__ dst,
                    const float* __restrict__ el, const float* __restrict__ er,
                    const unsigned int* __restrict__ nmax,
                    float* __restrict__ denom) {
    int e = blockIdx.x * blockDim.x + threadIdx.x;
    if (e >= N_EDGES) return;
    int s = src[e], d = dst[e];
    #pragma unroll
    for (int hh = 0; hh < HEADS; ++hh) {
        float v  = lrelu(el[s * HEADS + hh] + er[d * HEADS + hh]);
        float m  = decF(nmax[d * HEADS + hh]);
        float ex = __expf(v - m);
        atomicAdd(&denom[d * HEADS + hh], ex);
    }
}

// ---------------- kernel 5: alpha * h[src] scattered into out -------------
// One thread per (edge, head): 4 float4 gathers of h (L2-resident) + 16 atomics.
__global__ __launch_bounds__(256)
void gat_edge_aggregate(const int* __restrict__ src, const int* __restrict__ dst,
                        const float* __restrict__ el, const float* __restrict__ er,
                        const unsigned int* __restrict__ nmax,
                        const float* __restrict__ denom,
                        const float* __restrict__ h, float* __restrict__ out) {
    int i = blockIdx.x * blockDim.x + threadIdx.x;     // edge*HEADS + head
    if (i >= N_EDGES * HEADS) return;
    int e = i >> 2, hh = i & 3;
    int s = src[e], d = dst[e];
    float v  = lrelu(el[s * HEADS + hh] + er[d * HEADS + hh]);
    float m  = decF(nmax[d * HEADS + hh]);
    float dn = denom[d * HEADS + hh];
    float alpha = __expf(v - m) / dn;

    const float* hp = h + (size_t)s * HD + hh * OUT_DIM;
    float* op = out + (size_t)d * HD + hh * OUT_DIM;
    #pragma unroll
    for (int q = 0; q < 4; ++q) {
        float4 hv = *(const float4*)(hp + q * 4);
        atomicAdd(&op[q * 4 + 0], hv.x * alpha);
        atomicAdd(&op[q * 4 + 1], hv.y * alpha);
        atomicAdd(&op[q * 4 + 2], hv.z * alpha);
        atomicAdd(&op[q * 4 + 3], hv.w * alpha);
    }
}

// ---------------------------------------------------------------------------
extern "C" void kernel_launch(void* const* d_in, const int* in_sizes, int n_in,
                              void* d_out, int out_size, void* d_ws, size_t ws_size,
                              hipStream_t stream) {
    const float* feat   = (const float*)d_in[0];   // [N, 128]
    const float* W      = (const float*)d_in[1];   // [128, 64]
    const float* attn_l = (const float*)d_in[2];   // [4, 16]
    const float* attn_r = (const float*)d_in[3];   // [4, 16]
    const float* bias   = (const float*)d_in[4];   // [64]
    const int*   src    = (const int*)d_in[5];     // [E]
    const int*   dst    = (const int*)d_in[6];     // [E]
    float* out = (float*)d_out;                    // [N, 4, 16]

    // workspace layout
    char* ws = (char*)d_ws;
    float*        h     = (float*)ws;                              ws += (size_t)N_NODES * HD * sizeof(float);     // 25.6 MB
    float*        el    = (float*)ws;                              ws += (size_t)N_NODES * HEADS * sizeof(float);  // 1.6 MB
    float*        er    = (float*)ws;                              ws += (size_t)N_NODES * HEADS * sizeof(float);
    unsigned int* nmax  = (unsigned int*)ws;                       ws += (size_t)N_NODES * HEADS * sizeof(unsigned int);
    float*        denom = (float*)ws;

    const int T = 256;

    gat_init<<<(N_NODES * HD + T - 1) / T, T, 0, stream>>>(out, bias, denom, nmax);

    gat_gemm_wmma<<<N_NODES / 16, 128, 0, stream>>>(feat, W, h);

    gat_scores<<<(N_NODES * HEADS + T - 1) / T, T, 0, stream>>>(h, attn_l, attn_r, el, er);

    gat_edge_max<<<(N_EDGES + T - 1) / T, T, 0, stream>>>(src, dst, el, er, nmax);

    gat_edge_denom<<<(N_EDGES + T - 1) / T, T, 0, stream>>>(src, dst, el, er, nmax, denom);

    gat_edge_aggregate<<<((size_t)N_EDGES * HEADS + T - 1) / T, T, 0, stream>>>(
        src, dst, el, er, nmax, denom, h, out);
}
